// Qwen3MoeSparseMoeBlock_89833535963571
// MI455X (gfx1250) — compile-verified, matrix operand order checked by reading, not run on hardware
//
#include <hip/hip_runtime.h>
#include <hip/hip_bf16.h>
#include <stdint.h>

#define TOKS 512
#define HID  2048
#define NEXP 64
#define TOPK 8
#define ISZ  768

typedef __attribute__((ext_vector_type(16))) __bf16 v16bf;
typedef __attribute__((ext_vector_type(8)))  float  v8f;

union BF16Frag { v16bf v; uint32_t u[8]; };

// One-instruction f32x2 -> bf16x2 (truncate): v_perm_b32 picks the high 16
// bits of each source dword.  result = { hi16(b), hi16(a) }.
__device__ __forceinline__ uint32_t pack_bf16x2(float a, float b) {
  union { float f; uint32_t u; } ua, ub;
  ua.f = a; ub.f = b;
  return __builtin_amdgcn_perm(ub.u, ua.u, 0x07060302u);
}

__device__ __forceinline__ uint16_t f32_to_bf16(float a) {
  union { float f; uint32_t u; } ua; ua.f = a;
  return (uint16_t)(ua.u >> 16);
}

// WMMA bf16 A/B fragment K layout per lane: 8 consecutive K at (kb + khalf*8),
// then 8 consecutive K at (kb + khalf*8 + 16).
__device__ __forceinline__ void load_b_f32(const float* row, int kb, int khalf, BF16Frag& f) {
  const float4* p0 = (const float4*)(row + kb + khalf * 8);
  const float4* p1 = (const float4*)(row + kb + khalf * 8 + 16);
  float4 a0 = p0[0], a1 = p0[1];
  float4 b0 = p1[0], b1 = p1[1];
  f.u[0] = pack_bf16x2(a0.x, a0.y); f.u[1] = pack_bf16x2(a0.z, a0.w);
  f.u[2] = pack_bf16x2(a1.x, a1.y); f.u[3] = pack_bf16x2(a1.z, a1.w);
  f.u[4] = pack_bf16x2(b0.x, b0.y); f.u[5] = pack_bf16x2(b0.z, b0.w);
  f.u[6] = pack_bf16x2(b1.x, b1.y); f.u[7] = pack_bf16x2(b1.z, b1.w);
}

__device__ __forceinline__ void load_a_lds(const uint32_t* rowu32, int kb, int khalf, BF16Frag& f) {
  int c0 = (kb + khalf * 8) >> 1;          // 16B aligned (kb%32==0, khalf*8/2=4 u32)
  uint4 x0 = *(const uint4*)(rowu32 + c0);
  uint4 x1 = *(const uint4*)(rowu32 + c0 + 8);
  f.u[0] = x0.x; f.u[1] = x0.y; f.u[2] = x0.z; f.u[3] = x0.w;
  f.u[4] = x1.x; f.u[5] = x1.y; f.u[6] = x1.z; f.u[7] = x1.w;
}

// ---------------- kernel 0: zero output accumulator + expert counts ----------------
__global__ void moe_init(float* __restrict__ out_final, int* __restrict__ counts) {
  int i = blockIdx.x * blockDim.x + threadIdx.x;
  if (i < TOKS * HID) out_final[i] = 0.f;
  if (i < NEXP) counts[i] = 0;
}

// ---------------- kernel 1: router (logits + softmax + top-8 scatter) ----------------
__global__ __launch_bounds__(64) void moe_router(const float* __restrict__ x,
                                                 const float* __restrict__ gw,
                                                 float* __restrict__ logits_out,
                                                 int* __restrict__ counts,
                                                 int* __restrict__ idx_list,
                                                 float* __restrict__ cw_list) {
  __shared__ float lg[NEXP];
  __shared__ float pr[NEXP];
  const int t = blockIdx.x;
  const int e = threadIdx.x;
  const float4* xr = (const float4*)(x + (size_t)t * HID);
  const float4* gr = (const float4*)(gw + (size_t)e * HID);
  float acc = 0.f;
#pragma unroll 4
  for (int i = 0; i < HID / 4; ++i) {
    float4 a = xr[i], b = gr[i];
    acc += a.x * b.x + a.y * b.y + a.z * b.z + a.w * b.w;
  }
  lg[e] = acc;
  logits_out[(size_t)t * NEXP + e] = acc;
  __syncthreads();
  if (e == 0) {
    float mx = lg[0];
    for (int i = 1; i < NEXP; ++i) mx = fmaxf(mx, lg[i]);
    float s = 0.f;
    for (int i = 0; i < NEXP; ++i) { float p = __expf(lg[i] - mx); pr[i] = p; s += p; }
    float inv = __builtin_amdgcn_rcpf(s);
    int ids[TOPK]; float ws[TOPK];
    float tsum = 0.f;
    for (int k = 0; k < TOPK; ++k) {
      int bi = 0; float bv = -1.f;
      for (int i = 0; i < NEXP; ++i) { if (pr[i] > bv) { bv = pr[i]; bi = i; } }
      ids[k] = bi; ws[k] = bv * inv; pr[bi] = -2.f; tsum += ws[k];
    }
    float itn = __builtin_amdgcn_rcpf(tsum);
    for (int k = 0; k < TOPK; ++k) {
      int ex = ids[k];
      int pos = atomicAdd(&counts[ex], 1);
      idx_list[ex * TOKS + pos] = t;
      cw_list[ex * TOKS + pos] = ws[k] * itn;
    }
  }
}

// ---------------- kernel 2: fused expert FFN (bf16 WMMA), sparse token tiles ----------------
__global__ __launch_bounds__(256) void moe_expert(const float* __restrict__ x,
                                                  const float* __restrict__ w13,
                                                  const float* __restrict__ w2,
                                                  const int* __restrict__ counts,
                                                  const int* __restrict__ idx_list,
                                                  const float* __restrict__ cw_list,
                                                  float* __restrict__ out) {
  __shared__ uint32_t xs32[16][HID / 2];   // 64 KB: 16-token x tile in bf16
  __shared__ uint32_t act32[16][ISZ / 2];  // 24 KB: bf16 activations
  __shared__ int   toks[16];
  __shared__ float cwt[16];

  const int e    = blockIdx.y;
  const int cnt  = counts[e];
  const int tile = blockIdx.x;
  if (tile * 16 >= cnt) return;            // uniform over the block

  const int tid = threadIdx.x;
  if (tid < 16) {
    int slot = tile * 16 + tid;
    if (slot < cnt) { toks[tid] = idx_list[e * TOKS + slot]; cwt[tid] = cw_list[e * TOKS + slot]; }
    else            { toks[tid] = idx_list[e * TOKS];        cwt[tid] = 0.f; }   // padded row
  }
  __syncthreads();

  // ---- stage 0: x tile -> LDS (f32 -> bf16, one v_perm per pair) ----
  {
    int row = tid >> 4;                    // 0..15
    int seg = tid & 15;                    // each seg covers 128 floats
    const float4* xrow = (const float4*)(x + (size_t)toks[row] * HID + seg * 128);
    uint2* dst = (uint2*)&xs32[row][seg * 64];
#pragma unroll
    for (int i = 0; i < 32; ++i) {
      float4 v = xrow[i];
      uint2 p;
      p.x = pack_bf16x2(v.x, v.y);
      p.y = pack_bf16x2(v.z, v.w);
      dst[i] = p;
    }
  }
  __syncthreads();

  const int wave   = tid >> 5;
  const int lane   = tid & 31;
  const int lane16 = lane & 15;
  const int khalf  = lane >> 4;

  // ---- stage 1: [16,H] x w13_e^T -> silu(g)*u -> act[16,768] bf16 in LDS ----
  uint16_t* actu = (uint16_t*)act32;
  for (int jb = wave; jb < ISZ / 16; jb += 8) {
    const int j = jb * 16 + lane16;        // output feature handled by this lane
    const float* wg = w13 + ((size_t)e * 2 * ISZ + j) * HID;
    const float* wu = w13 + ((size_t)e * 2 * ISZ + ISZ + j) * HID;
    v8f cg = {}; v8f cu = {};
#pragma unroll 2
    for (int kb = 0; kb < HID; kb += 32) {
      if ((kb & 63) == 0) {
        __builtin_prefetch(wg + kb + 512, 0, 1);
        __builtin_prefetch(wu + kb + 512, 0, 1);
      }
      BF16Frag a, bg, bu;
      load_a_lds(xs32[lane16], kb, khalf, a);
      load_b_f32(wg, kb, khalf, bg);
      load_b_f32(wu, kb, khalf, bu);
      cg = __builtin_amdgcn_wmma_f32_16x16x32_bf16(false, a.v, false, bg.v, (short)0, cg, false, false);
      cu = __builtin_amdgcn_wmma_f32_16x16x32_bf16(false, a.v, false, bu.v, (short)0, cu, false, false);
    }
#pragma unroll
    for (int r = 0; r < 8; ++r) {
      int m = r + khalf * 8;               // D layout: lane n=lane16, rows m=r+khalf*8
      float g = cg[r];
      float sig = __builtin_amdgcn_rcpf(1.f + __expf(-g));   // fast sigmoid
      actu[m * ISZ + j] = f32_to_bf16(g * sig * cu[r]);
    }
  }
  __syncthreads();

  // ---- stage 2: act[16,768] x w2_e^T -> scaled atomic accumulate into out ----
  for (int nb = wave; nb < HID / 16; nb += 8) {
    const int n = nb * 16 + lane16;
    const float* wr = w2 + ((size_t)e * HID + n) * ISZ;
    v8f acc = {};
#pragma unroll 2
    for (int kb = 0; kb < ISZ; kb += 32) {
      if ((kb & 63) == 0) __builtin_prefetch(wr + kb + 256, 0, 1);
      BF16Frag a, b;
      load_a_lds(act32[lane16], kb, khalf, a);
      load_b_f32(wr, kb, khalf, b);
      acc = __builtin_amdgcn_wmma_f32_16x16x32_bf16(false, a.v, false, b.v, (short)0, acc, false, false);
    }
#pragma unroll
    for (int r = 0; r < 8; ++r) {
      int m = r + khalf * 8;
      float c = cwt[m];
      if (c != 0.f) atomicAdd(&out[(size_t)toks[m] * HID + n], c * acc[r]);
    }
  }
}

extern "C" void kernel_launch(void* const* d_in, const int* in_sizes, int n_in,
                              void* d_out, int out_size, void* d_ws, size_t ws_size,
                              hipStream_t stream) {
  (void)in_sizes; (void)n_in; (void)out_size; (void)ws_size;
  const float* x    = (const float*)d_in[0];  // [1,512,2048]
  const float* gw   = (const float*)d_in[1];  // [64,2048]
  const float* w13  = (const float*)d_in[2];  // [64,1536,2048]
  const float* w2   = (const float*)d_in[3];  // [64,2048,768]
  float* out    = (float*)d_out;                       // final [512*2048]
  float* logits = out + (size_t)TOKS * HID;            // router_logits [512*64]

  int*   counts   = (int*)d_ws;                        // [64]
  int*   idx_list = counts + NEXP;                     // [64*512]
  float* cw_list  = (float*)(idx_list + NEXP * TOKS);  // [64*512]

  moe_init<<<(TOKS * HID + 255) / 256, 256, 0, stream>>>(out, counts);
  moe_router<<<TOKS, NEXP, 0, stream>>>(x, gw, logits, counts, idx_list, cw_list);
  dim3 grid(TOKS / 16, NEXP);
  moe_expert<<<grid, 256, 0, stream>>>(x, w13, w2, counts, idx_list, cw_list, out);
}
